// AdaptiveLocalConv_38955353375517
// MI455X (gfx1250) — matrix operand order repaired and checked by reading, not compiled, not run on hardware
//
#include <hip/hip_runtime.h>
#include <hip/hip_bf16.h>

// ---------------- problem constants ----------------
constexpr int B_ = 2, L_ = 4096, C_ = 512, H_ = 8, K_ = 64, D_ = 64;
constexpr int HID_ = 128;            // C/4
constexpr int MAXW_ = 12;            // int(L**0.3) clipped to K
constexpr int HALFMAX_ = 6;          // MAXW/2
constexpr int S_ = 13;               // 2*HALFMAX+1
constexpr float MAXOFF_ = 12.0f;     // float(int(L**0.3))
constexpr int KD_ = 512;             // GEMM K dim (all three GEMMs)

// ---------------- WMMA types ----------------
typedef __attribute__((ext_vector_type(16))) __bf16 v16bf;
typedef __attribute__((ext_vector_type(8)))  float  v8f;
struct U4x2 { uint4 lo, hi; };

__device__ __forceinline__ unsigned short f32_to_bf16(float f) {
  unsigned int u = __float_as_uint(f);
  unsigned int r = u + 0x7FFFu + ((u >> 16) & 1u);   // round-to-nearest-even
  return (unsigned short)(r >> 16);
}
__device__ __forceinline__ float sigmoidf_(float x) { return 1.0f / (1.0f + __expf(-x)); }
__device__ __forceinline__ float siluf_(float x)    { return x * sigmoidf_(x); }

// A fragment 16x32 bf16: per-lane two contiguous 8-elem runs (ISA 7.12.2 layout)
__device__ __forceinline__ v16bf load_afrag(const unsigned short* p) {
  U4x2 t; t.lo = *(const uint4*)(p); t.hi = *(const uint4*)(p + 16);
  return __builtin_bit_cast(v16bf, t);
}
// B fragment 32x16 bf16: per-lane two contiguous 8-elem runs (same k order as A)
__device__ __forceinline__ v16bf load_bfrag(const unsigned short* p) {
  U4x2 t; t.lo = *(const uint4*)(p); t.hi = *(const uint4*)(p + 16);
  return __builtin_bit_cast(v16bf, t);
}

// ---------------- f32 -> bf16 conversion ----------------
__global__ __launch_bounds__(256)
void cvt_bf16_kernel(const float* __restrict__ src, unsigned short* __restrict__ dst, int n) {
  int i = blockIdx.x * 256 + threadIdx.x;
  if (i < n) dst[i] = f32_to_bf16(src[i]);
}

// ---------------- window / offset tiny projections ----------------
__global__ __launch_bounds__(256)
void winoff_kernel(const float* __restrict__ x,
                   const float* __restrict__ ww, const float* __restrict__ wb, const float* __restrict__ wg,
                   const float* __restrict__ ow, const float* __restrict__ ob, const float* __restrict__ og,
                   float* __restrict__ halfwin, float* __restrict__ centeroff) {
  __shared__ float sW[H_], sO[H_];
  const int pos  = blockIdx.x;            // b*L + l
  const int h    = threadIdx.x >> 5;      // 8 waves = 8 heads
  const int lane = threadIdx.x & 31;
  const float* xr  = x  + (size_t)pos * C_;
  const float* wr  = ww + (size_t)h * C_;
  const float* orr = ow + (size_t)h * C_;
  float dw = 0.f, dofs = 0.f;
  for (int c = lane; c < C_; c += 32) { float xv = xr[c]; dw += xv * wr[c]; dofs += xv * orr[c]; }
  for (int o = 16; o > 0; o >>= 1) { dw += __shfl_down(dw, o); dofs += __shfl_down(dofs, o); }
  if (lane == 0) { sW[h] = dw + wb[h]; sO[h] = dofs + ob[h]; }
  __syncthreads();
  if (threadIdx.x < H_) {
    const int t = threadIdx.x;
    float ssw = 0.f, sso = 0.f;
    #pragma unroll
    for (int i = 0; i < H_; ++i) { ssw += sW[i] * sW[i]; sso += sO[i] * sO[i]; }
    const float rsw = rsqrtf(ssw * (1.0f / H_) + 1e-6f);
    const float rso = rsqrtf(sso * (1.0f / H_) + 1e-6f);
    const float raw = sigmoidf_(wg[t] * (sW[t] * rsw));
    const float wsz = 1.0f + raw * ((float)MAXW_ - 1.0f);
    halfwin[(size_t)pos * H_ + t]   = fmaxf(wsz * 0.5f, 0.5f);
    centeroff[(size_t)pos * H_ + t] = tanhf(og[t] * (sO[t] * rso)) * MAXOFF_;
  }
}

// ---------------- bf16 WMMA GEMM: Cout[m,n] = sum_k A[m,k]*W[n,k] (+bias) (+silu) ----------------
// One wave per 32x32 output tile: 2 A-frags x 2 B-frags -> 4 accumulators.
// K hardcoded to 512: 16 unrolled k-steps x 4 wmma = 64 wmma per wave,
// 8 b128 loads per 4 wmma (2 vmem per wmma).
__global__ __launch_bounds__(256)
void gemm_bf16_kernel(const unsigned short* __restrict__ A,
                      const unsigned short* __restrict__ W,
                      const float* __restrict__ bias,
                      float* __restrict__ Cout,
                      int M, int N, int epilogue) {
  const int lane = threadIdx.x & 31;
  const int wave = threadIdx.x >> 5;
  const int half = lane >> 4;
  const int r    = lane & 15;
  const int tilesN = N >> 5;                        // 32-wide tiles
  const int tile = blockIdx.x * 8 + wave;
  if (tile >= (M >> 5) * tilesN) return;
  const int m0 = (tile / tilesN) << 5;
  const int n0 = (tile % tilesN) << 5;

  const unsigned short* a0p = A + (size_t)(m0 + r) * KD_ + half * 8;        // rows m0..m0+15
  const unsigned short* a1p = a0p + (size_t)16 * KD_;                        // rows m0+16..m0+31
  const unsigned short* b0p = W + (size_t)(n0 + r) * KD_ + half * 8;         // cols n0..n0+15
  const unsigned short* b1p = b0p + (size_t)16 * KD_;                        // cols n0+16..n0+31

  v8f acc00 = {}, acc01 = {}, acc10 = {}, acc11 = {};
  #pragma unroll
  for (int kk = 0; kk < KD_ / 32; ++kk) {
    v16bf a0 = load_afrag(a0p);
    v16bf a1 = load_afrag(a1p);
    v16bf b0 = load_bfrag(b0p);
    v16bf b1 = load_bfrag(b1p);
    __builtin_prefetch(a0p + 64, 0, 2);
    __builtin_prefetch(b0p + 64, 0, 2);
    a0p += 32; a1p += 32; b0p += 32; b1p += 32;
    acc00 = __builtin_amdgcn_wmma_f32_16x16x32_bf16(false, a0, false, b0, (short)0, acc00, false, false);
    acc01 = __builtin_amdgcn_wmma_f32_16x16x32_bf16(false, a0, false, b1, (short)0, acc01, false, false);
    acc10 = __builtin_amdgcn_wmma_f32_16x16x32_bf16(false, a1, false, b0, (short)0, acc10, false, false);
    acc11 = __builtin_amdgcn_wmma_f32_16x16x32_bf16(false, a1, false, b1, (short)0, acc11, false, false);
  }
  const float bv0 = bias ? bias[n0 + r]      : 0.0f;
  const float bv1 = bias ? bias[n0 + 16 + r] : 0.0f;
  #pragma unroll
  for (int j = 0; j < 8; ++j) {
    const int row0 = m0 + half * 8 + j;          // D layout: VGPR j, lanes<16 -> M=j, lanes>=16 -> M=8+j
    const int row1 = row0 + 16;
    float v00 = acc00[j] + bv0, v01 = acc01[j] + bv1;
    float v10 = acc10[j] + bv0, v11 = acc11[j] + bv1;
    if (epilogue == 1) { v00 = siluf_(v00); v01 = siluf_(v01); v10 = siluf_(v10); v11 = siluf_(v11); }
    Cout[(size_t)row0 * N + (n0 + r)]      = v00;
    Cout[(size_t)row0 * N + (n0 + 16 + r)] = v01;
    Cout[(size_t)row1 * N + (n0 + r)]      = v10;
    Cout[(size_t)row1 * N + (n0 + 16 + r)] = v11;
  }
}

// ---------------- in-place rmsnorm(width=512) + silu ----------------
__global__ __launch_bounds__(256)
void rmsnorm_silu_kernel(float* __restrict__ buf, const float* __restrict__ gamma) {
  __shared__ float red[8];
  float* row = buf + (size_t)blockIdx.x * C_;
  const float v0 = row[threadIdx.x], v1 = row[threadIdx.x + 256];
  float ss = v0 * v0 + v1 * v1;
  for (int o = 16; o > 0; o >>= 1) ss += __shfl_down(ss, o);
  if ((threadIdx.x & 31) == 0) red[threadIdx.x >> 5] = ss;
  __syncthreads();
  float tot = 0.f;
  #pragma unroll
  for (int i = 0; i < 8; ++i) tot += red[i];
  const float rs = rsqrtf(tot * (1.0f / C_) + 1e-6f);
  const float y0 = gamma[threadIdx.x] * v0 * rs;
  const float y1 = gamma[threadIdx.x + 256] * v1 * rs;
  row[threadIdx.x]       = siluf_(y0);
  row[threadIdx.x + 256] = siluf_(y1);
}

// ---------------- deformable local gather: one wave per (b,l,h) ----------------
__global__ __launch_bounds__(256)
void gather_kernel(const float* __restrict__ v, const float* __restrict__ kact,
                   const float* __restrict__ halfwin, const float* __restrict__ centeroff,
                   float* __restrict__ outh) {
  const int wave = threadIdx.x >> 5, lane = threadIdx.x & 31;
  const int gid = blockIdx.x * 8 + wave;        // flat (b,l,h)
  const int b = gid / (L_ * H_);
  const int rem = gid % (L_ * H_);
  const int l = rem / H_, h = rem % H_;
  const float hw = halfwin[gid];
  const float co = centeroff[gid];
  const float* krow = kact + (size_t)gid * K_;
  const int d0 = lane * 2;
  float a0 = 0.f, a1 = 0.f, wsum = 0.f;
  #pragma unroll
  for (int s = 0; s < S_; ++s) {
    const float so  = (float)(s - HALFMAX_);
    const float rel = fabsf(so) / hw;
    const float winw = __expf(-rel * rel);
    const float np = fminf(rel, 1.0f) * (float)(K_ - 1);
    int idx = (int)np; if (idx > K_ - 2) idx = K_ - 2;
    const float wc = np - (float)idx;
    const float kw = fmaxf(krow[idx] * (1.0f - wc) + krow[idx + 1] * wc, 0.0f);
    const float nb = (float)l + co + so;
    const float valid = (nb >= 0.0f && nb < (float)L_) ? 1.0f : 0.0f;
    const float wgt = (kw + 1.0f) * winw * valid;
    wsum += wgt;
    const float pc = fminf(fmaxf(nb, 0.0f), (float)L_ - 1.001f);
    int pf = (int)floorf(pc); pf = pf < 0 ? 0 : (pf > L_ - 1 ? L_ - 1 : pf);
    const int pcl = (pf + 1 > L_ - 1) ? L_ - 1 : pf + 1;
    const float fr = pc - (float)pf;
    const float2 vf = *(const float2*)(v + ((size_t)(b * L_ + pf)  * H_ + h) * D_ + d0);
    const float2 vc = *(const float2*)(v + ((size_t)(b * L_ + pcl) * H_ + h) * D_ + d0);
    a0 += wgt * (vf.x * (1.0f - fr) + vc.x * fr);
    a1 += wgt * (vf.y * (1.0f - fr) + vc.y * fr);
  }
  const float inv = 1.0f / fmaxf(wsum, 1.0f);
  float2 o; o.x = a0 * inv; o.y = a1 * inv;
  *(float2*)(outh + (size_t)gid * D_ + d0) = o;
}

// ---------------- mean over L per (b,c) ----------------
__global__ __launch_bounds__(256)
void mean_kernel(const float* __restrict__ outh, float* __restrict__ mean) {
  __shared__ float red[8];
  const int b = blockIdx.x >> 9;      // C_=512 columns
  const int c = blockIdx.x & 511;
  float s = 0.f;
  for (int l = threadIdx.x; l < L_; l += 256) s += outh[((size_t)(b * L_ + l)) * C_ + c];
  for (int o = 16; o > 0; o >>= 1) s += __shfl_down(s, o);
  if ((threadIdx.x & 31) == 0) red[threadIdx.x >> 5] = s;
  __syncthreads();
  if (threadIdx.x == 0) {
    float tot = 0.f;
    #pragma unroll
    for (int i = 0; i < 8; ++i) tot += red[i];
    mean[b * C_ + c] = tot * (1.0f / L_);
  }
}

// ---------------- squeeze-excite MLP (one block per batch) ----------------
__global__ __launch_bounds__(256)
void se_kernel(const float* __restrict__ mean, const float* __restrict__ w1,
               const float* __restrict__ w2, float* __restrict__ scale) {
  __shared__ float m[C_];
  __shared__ float hbuf[HID_];
  const int b = blockIdx.x;
  m[threadIdx.x]       = mean[b * C_ + threadIdx.x];
  m[threadIdx.x + 256] = mean[b * C_ + threadIdx.x + 256];
  __syncthreads();
  if (threadIdx.x < HID_) {
    const float* wr = w1 + (size_t)threadIdx.x * C_;
    float d = 0.f;
    for (int c = 0; c < C_; ++c) d += wr[c] * m[c];
    hbuf[threadIdx.x] = siluf_(d);
  }
  __syncthreads();
  for (int c = threadIdx.x; c < C_; c += 256) {
    const float* wr = w2 + (size_t)c * HID_;
    float d = 0.f;
    for (int hh = 0; hh < HID_; ++hh) d += wr[hh] * hbuf[hh];
    scale[b * C_ + c] = sigmoidf_(d);
  }
}

// ---------------- scale by SE + convert to bf16 for final GEMM ----------------
__global__ __launch_bounds__(256)
void scale_bf16_kernel(const float* __restrict__ outh, const float* __restrict__ scale,
                       unsigned short* __restrict__ dst) {
  const size_t i = (size_t)blockIdx.x * 256 + threadIdx.x;
  const int c = (int)(i & (C_ - 1));
  const int b = (int)(i >> 21);                 // L*C = 2^21
  dst[i] = f32_to_bf16(outh[i] * scale[b * C_ + c]);
}

// ---------------- host launcher ----------------
extern "C" void kernel_launch(void* const* d_in, const int* in_sizes, int n_in,
                              void* d_out, int out_size, void* d_ws, size_t ws_size,
                              hipStream_t stream) {
  const float* x   = (const float*)d_in[0];
  const float* window_w = (const float*)d_in[1];
  const float* window_b = (const float*)d_in[2];
  const float* window_g = (const float*)d_in[3];
  const float* offset_w = (const float*)d_in[4];
  const float* offset_b = (const float*)d_in[5];
  const float* offset_g = (const float*)d_in[6];
  const float* kernel_w = (const float*)d_in[7];
  const float* kernel_b = (const float*)d_in[8];
  const float* kernel_g = (const float*)d_in[9];
  const float* v_w = (const float*)d_in[10];
  const float* v_b = (const float*)d_in[11];
  const float* se_w1 = (const float*)d_in[12];
  const float* se_w2 = (const float*)d_in[13];
  const float* out_w = (const float*)d_in[14];
  float* out = (float*)d_out;

  const size_t BLC = (size_t)B_ * L_ * C_;      // 4,194,304
  const size_t CC  = (size_t)C_ * C_;           // 262,144
  const size_t BLH = (size_t)B_ * L_ * H_;      // 65,536

  // workspace carve-up (256B aligned)
  char* ws = (char*)d_ws;
  size_t off = 0;
  auto carve = [&](size_t bytes) { char* p = ws + off; off = (off + bytes + 255) & ~(size_t)255; return p; };
  unsigned short* xb   = (unsigned short*)carve(BLC * 2);   // x in bf16 (reused as out_scaled bf16)
  unsigned short* kwb  = (unsigned short*)carve(CC * 2);
  unsigned short* vwb  = (unsigned short*)carve(CC * 2);
  unsigned short* owb  = (unsigned short*)carve(CC * 2);
  float* kraw  = (float*)carve(BLC * 4);                    // kernel proj -> rmsnorm+silu in place
  float* vbuf  = (float*)carve(BLC * 4);                    // v projection
  float* winb  = (float*)carve(BLH * 4);                    // half_win
  float* offb  = (float*)carve(BLH * 4);                    // center_off
  float* outh  = (float*)carve(BLC * 4);                    // gathered heads
  float* meanb = (float*)carve((size_t)B_ * C_ * 4);
  float* scaleb= (float*)carve((size_t)B_ * C_ * 4);
  (void)ws_size; (void)n_in; (void)in_sizes; (void)out_size;

  // 1) convert x and the three 512x512 weights to bf16
  cvt_bf16_kernel<<<(int)(BLC / 256), 256, 0, stream>>>(x, xb, (int)BLC);
  cvt_bf16_kernel<<<(int)(CC / 256), 256, 0, stream>>>(kernel_w, kwb, (int)CC);
  cvt_bf16_kernel<<<(int)(CC / 256), 256, 0, stream>>>(v_w, vwb, (int)CC);
  cvt_bf16_kernel<<<(int)(CC / 256), 256, 0, stream>>>(out_w, owb, (int)CC);

  // 2) window/offset tiny projections (one block per position)
  winoff_kernel<<<B_ * L_, 256, 0, stream>>>(x, window_w, window_b, window_g,
                                             offset_w, offset_b, offset_g, winb, offb);

  // 3) WMMA GEMMs: kernel proj and v proj (M=8192, N=512, K=512), 32x32 tile/wave
  const int gemmBlocks = (B_ * L_ / 32) * (C_ / 32) / 8;    // 512
  gemm_bf16_kernel<<<gemmBlocks, 256, 0, stream>>>(xb, kwb, kernel_b, kraw, B_ * L_, C_, 0);
  gemm_bf16_kernel<<<gemmBlocks, 256, 0, stream>>>(xb, vwb, v_b,      vbuf, B_ * L_, C_, 0);

  // 4) rmsnorm + silu on kernel projection (in place, width 512)
  rmsnorm_silu_kernel<<<B_ * L_, 256, 0, stream>>>(kraw, kernel_g);

  // 5) deformable gather (one wave per (b,l,h))
  gather_kernel<<<(int)(BLH / 8), 256, 0, stream>>>(vbuf, kraw, winb, offb, outh);

  // 6) channel means over L
  mean_kernel<<<B_ * C_, 256, 0, stream>>>(outh, meanb);

  // 7) squeeze-excite MLP
  se_kernel<<<B_, 256, 0, stream>>>(meanb, se_w1, se_w2, scaleb);

  // 8) apply SE scale, convert to bf16 (reuse xb slot)
  scale_bf16_kernel<<<(int)(BLC / 256), 256, 0, stream>>>(outh, scaleb, xb);

  // 9) final WMMA GEMM with silu epilogue -> d_out (f32)
  gemm_bf16_kernel<<<gemmBlocks, 256, 0, stream>>>(xb, owb, nullptr, out, B_ * L_, C_, 1);
}
